// SummaRuNNer_29102698398258
// MI455X (gfx1250) — matrix operand order repaired and checked by reading, not compiled, not run on hardware
//
#include <hip/hip_runtime.h>
#include <hip/hip_bf16.h>

// SummaRuNNer forward for MI455X (gfx1250, wave32, WMMA).
// B=64, L=256, E=512, H=512, 2H=1024, 4H=2048.

typedef __attribute__((ext_vector_type(16))) __bf16 v16bf;
typedef __attribute__((ext_vector_type(8)))  __bf16 v8bf;
typedef __attribute__((ext_vector_type(8)))  float  v8f;

#define Bn   64
#define Ln   256
#define En   512
#define Hh   512
#define G4   2048
#define D2   1024
#define ASTR 1040   // LDS row stride in bf16 elems (2080B, 16B aligned, bank-skewed)

__device__ __forceinline__ float sigmoidf_(float x) {
    return 1.0f / (1.0f + __expf(-x));
}

// ---------------------------------------------------------------------------
// Conversion / packing kernels (fp32 -> bf16)
// ---------------------------------------------------------------------------
__global__ void cvt_x_kernel(const float* __restrict__ x, __bf16* __restrict__ dst, int n) {
    int i = blockIdx.x * blockDim.x + threadIdx.x;
    if (i < n) dst[i] = (__bf16)x[i];
}

// Wcat[g, 0:512] = Wih[g,:], Wcat[g, 512:1024] = Whh[g,:]  (row-major [2048,1024])
__global__ void cvt_wcat_kernel(const float* __restrict__ Wih, const float* __restrict__ Whh,
                                __bf16* __restrict__ dst) {
    int i = blockIdx.x * blockDim.x + threadIdx.x;   // 2048*1024
    if (i >= G4 * D2) return;
    int g = i >> 10, k = i & 1023;
    float v = (k < 512) ? Wih[(size_t)g * 512 + k] : Whh[(size_t)g * 512 + (k - 512)];
    dst[i] = (__bf16)v;
}

// WnvT[e, d] = W_nov[d, e]  (transpose so B-fragment reads are contiguous)
__global__ void cvt_wnovT_kernel(const float* __restrict__ Wnov, __bf16* __restrict__ dst) {
    int i = blockIdx.x * blockDim.x + threadIdx.x;   // 1024*1024, i = e*1024 + d
    if (i >= D2 * D2) return;
    int e = i >> 10, d = i & 1023;
    dst[i] = (__bf16)Wnov[(size_t)d * D2 + e];
}

// ---------------------------------------------------------------------------
// Bidirectional LSTM scan. grid = 8 blocks (dir*4 + batch-tile), block = 512.
// Each WG owns 16 batch rows; per step computes z = [x_l|h] @ Wcat^T + b via
// WMMA, applies gates, keeps h in LDS (bf16) and c in registers.
// ---------------------------------------------------------------------------
__global__ void __launch_bounds__(512)
lstm_kernel(const __bf16* __restrict__ xbf,
            const __bf16* __restrict__ WcatF, const __bf16* __restrict__ WcatB,
            const float* __restrict__ bF, const float* __restrict__ bB,
            const int* __restrict__ lens,
            float* __restrict__ Hout, __bf16* __restrict__ HoutBF) {
    __shared__ __bf16 Atile[16 * ASTR];   // [16 rows][1024 cols]: 0..511 x_l, 512..1023 h

    const int dir    = blockIdx.x >> 2;
    const int b_base = (blockIdx.x & 3) * 16;
    const __bf16* Wcat = dir ? WcatB : WcatF;
    const float*  bias = dir ? bB : bF;

    const int tid  = threadIdx.x;
    const int wave = tid >> 5;       // 0..15, owns h cols [wave*32, wave*32+32)
    const int lane = tid & 31;
    const int half = lane >> 4;
    const int n16  = lane & 15;
    const int m    = lane & 15;      // A-fragment row

    // Per-lane bias for the 4 gates x 2 N-tiles this wave owns.
    float bv[4][2];
#pragma unroll
    for (int gi = 0; gi < 4; ++gi)
#pragma unroll
        for (int t = 0; t < 2; ++t)
            bv[gi][t] = bias[gi * 512 + wave * 32 + t * 16 + n16];

    // Sequence lengths for the 8 C/D rows this lane covers.
    int lenr[8];
#pragma unroll
    for (int r = 0; r < 8; ++r) lenr[r] = lens[b_base + 8 * half + r];

    float cst[2][8];
#pragma unroll
    for (int t = 0; t < 2; ++t)
#pragma unroll
        for (int r = 0; r < 8; ++r) cst[t][r] = 0.0f;

    // Zero the h region of the A tile (h0 = 0).
    {
        int row = tid >> 5, c0 = 512 + (tid & 31) * 16;
#pragma unroll
        for (int j = 0; j < 16; ++j) Atile[row * ASTR + c0 + j] = (__bf16)0.0f;
    }
    __syncthreads();

    __builtin_prefetch(Wcat + (size_t)(wave * 32) * D2, 0, 3);

    for (int s = 0; s < Ln; ++s) {
        const int l = dir ? (Ln - 1 - s) : s;

        // --- Stage x_l tile (16 rows x 512 bf16) into LDS via async loads ---
        {
            int row = tid >> 5;
            int col0 = (tid & 31) * 16;           // 16 bf16 = 32B per thread
            const __bf16* gp = xbf + (((size_t)(b_base + row) * Ln + l) * En + col0);
            unsigned int ldsoff = (unsigned int)(size_t)(&Atile[row * ASTR + col0]);
            unsigned long long ga = (unsigned long long)(size_t)gp;
            asm volatile(
                "global_load_async_to_lds_b128 %0, %1, off\n\t"
                "global_load_async_to_lds_b128 %0, %1, off offset:16"
                :: "v"(ldsoff), "v"(ga) : "memory");
        }
        asm volatile("s_wait_asynccnt 0x0" ::: "memory");
        __syncthreads();   // x ready; previous step's h writes visible

        // --- z = A(16x1024) @ Wcat^T, this wave's 4 gates x 2 N-tiles ---
        v8f acc[4][2];
#pragma unroll
        for (int gi = 0; gi < 4; ++gi)
#pragma unroll
            for (int t = 0; t < 2; ++t)
#pragma unroll
                for (int r = 0; r < 8; ++r) acc[gi][t][r] = bv[gi][t];

        const int koff = half * 16;
        for (int kk = 0; kk < D2; kk += 32) {
            v8bf a0 = *(const v8bf*)&Atile[m * ASTR + kk + koff];
            v8bf a1 = *(const v8bf*)&Atile[m * ASTR + kk + koff + 8];
            v16bf a;
#pragma unroll
            for (int i = 0; i < 8; ++i) { a[i] = a0[i]; a[i + 8] = a1[i]; }
#pragma unroll
            for (int gi = 0; gi < 4; ++gi)
#pragma unroll
                for (int t = 0; t < 2; ++t) {
                    int g = gi * 512 + wave * 32 + t * 16 + n16;
                    v16bf bfr = *(const v16bf*)(Wcat + (size_t)g * D2 + kk + half * 16);
                    acc[gi][t] = __builtin_amdgcn_wmma_f32_16x16x32_bf16(
                        false, a, false, bfr, (short)0, acc[gi][t], false, false);
                }
        }
        __syncthreads();   // all A-tile reads done before h is overwritten

        // --- Gates, cell update, masked outputs ---
#pragma unroll
        for (int t = 0; t < 2; ++t) {
            int ncol = wave * 32 + t * 16 + n16;      // h column
#pragma unroll
            for (int r = 0; r < 8; ++r) {
                float iv = sigmoidf_(acc[0][t][r]);
                float fv = sigmoidf_(acc[1][t][r]);
                float gv = tanhf(acc[2][t][r]);
                float ov = sigmoidf_(acc[3][t][r]);
                float cn = fv * cst[t][r] + iv * gv;
                float hn = ov * tanhf(cn);
                int mrow = 8 * half + r;
                bool mk = l < lenr[r];
                float hval = mk ? hn : 0.0f;
                if (mk) {
                    cst[t][r] = cn;
                    Atile[mrow * ASTR + 512 + ncol] = (__bf16)hn;  // carry h
                }
                size_t orow = ((size_t)(b_base + mrow) * Ln + l) * D2 + (size_t)dir * 512 + ncol;
                Hout[orow]   = hval;
                HoutBF[orow] = (__bf16)hval;
            }
        }
    }
}

// ---------------------------------------------------------------------------
// Hn_pre[r, e] = Hout[r, :] @ W_nov (via transposed WnvT). grid = 1024, blk 512.
// ---------------------------------------------------------------------------
__global__ void __launch_bounds__(512)
hn_gemm_kernel(const __bf16* __restrict__ HoutBF, const __bf16* __restrict__ WnvT,
               float* __restrict__ Hn) {
    __shared__ __bf16 A2[16 * ASTR];
    const int r0   = blockIdx.x * 16;
    const int tid  = threadIdx.x;
    const int wave = tid >> 5;
    const int lane = tid & 31;
    const int half = lane >> 4;
    const int n16  = lane & 15;
    const int m    = lane & 15;

    // Stage 16x1024 bf16 A tile.
    {
        int row = tid >> 5, col0 = (tid & 31) * 32;
        const v8bf* src = (const v8bf*)(HoutBF + (size_t)(r0 + row) * D2 + col0);
        v8bf* dst = (v8bf*)&A2[row * ASTR + col0];
#pragma unroll
        for (int j = 0; j < 4; ++j) dst[j] = src[j];
    }
    __syncthreads();

    const int wbase = wave * 64;
    v8f acc[4];
#pragma unroll
    for (int t = 0; t < 4; ++t)
#pragma unroll
        for (int r = 0; r < 8; ++r) acc[t][r] = 0.0f;

    const int koff = half * 16;
    for (int kk = 0; kk < D2; kk += 32) {
        v8bf a0 = *(const v8bf*)&A2[m * ASTR + kk + koff];
        v8bf a1 = *(const v8bf*)&A2[m * ASTR + kk + koff + 8];
        v16bf a;
#pragma unroll
        for (int i = 0; i < 8; ++i) { a[i] = a0[i]; a[i + 8] = a1[i]; }
#pragma unroll
        for (int t = 0; t < 4; ++t) {
            int e = wbase + t * 16 + n16;
            v16bf bfr = *(const v16bf*)(WnvT + (size_t)e * D2 + kk + half * 16);
            acc[t] = __builtin_amdgcn_wmma_f32_16x16x32_bf16(
                false, a, false, bfr, (short)0, acc[t], false, false);
        }
    }
#pragma unroll
    for (int t = 0; t < 4; ++t) {
        int e = wbase + t * 16 + n16;
#pragma unroll
        for (int r = 0; r < 8; ++r)
            Hn[(size_t)(r0 + 8 * half + r) * D2 + e] = acc[t][r];
    }
}

// ---------------------------------------------------------------------------
// D[b,:] = (mean_l Hout[b,l,:]) @ W_doc^T + b_doc. grid = 64, blk 256.
// ---------------------------------------------------------------------------
__global__ void doc_kernel(const float* __restrict__ Hout, const int* __restrict__ dlen,
                           const float* __restrict__ Wdoc, const float* __restrict__ bdoc,
                           float* __restrict__ Dv) {
    __shared__ float dsum[D2];
    int b = blockIdx.x, tid = threadIdx.x;
    int len = dlen[b];
    float inv = 1.0f / (float)len;
    int e0 = tid * 4;
    float s0 = 0, s1 = 0, s2 = 0, s3 = 0;
    for (int l = 0; l < len; ++l) {
        const float* row = Hout + ((size_t)(b * Ln + l)) * D2 + e0;
        s0 += row[0]; s1 += row[1]; s2 += row[2]; s3 += row[3];
    }
    dsum[e0] = s0 * inv; dsum[e0 + 1] = s1 * inv; dsum[e0 + 2] = s2 * inv; dsum[e0 + 3] = s3 * inv;
    __syncthreads();
    int d0 = tid * 4;
#pragma unroll
    for (int j = 0; j < 4; ++j) {
        int d = d0 + j;
        float acc = bdoc[d];
        const float* wr = Wdoc + (size_t)d * D2;
        for (int e = 0; e < D2; ++e) acc += wr[e] * dsum[e];
        Dv[(size_t)b * D2 + d] = acc;
    }
}

// Tsal[b,d] = sum_e W_sal[d,e] * D[b,e]. grid = 64, blk 256.
__global__ void sal_kernel(const float* __restrict__ Wsal, const float* __restrict__ Dv,
                           float* __restrict__ Tsal) {
    __shared__ float drow[D2];
    int b = blockIdx.x, tid = threadIdx.x;
#pragma unroll
    for (int j = 0; j < 4; ++j) drow[tid * 4 + j] = Dv[(size_t)b * D2 + tid * 4 + j];
    __syncthreads();
#pragma unroll
    for (int j = 0; j < 4; ++j) {
        int d = tid * 4 + j;
        float acc = 0.0f;
        const float* wr = Wsal + (size_t)d * D2;
        for (int e = 0; e < D2; ++e) acc += wr[e] * drow[e];
        Tsal[(size_t)b * D2 + d] = acc;
    }
}

// base[b,l] = content + salience + position. grid = 2048, blk 256 (8 waves).
__global__ void base_kernel(const float* __restrict__ Hout, const float* __restrict__ wcon,
                            const float* __restrict__ bcon, const float* __restrict__ Tsal,
                            const float* __restrict__ bsal, const float* __restrict__ pos,
                            float* __restrict__ baseArr) {
    int wave = threadIdx.x >> 5, lane = threadIdx.x & 31;
    int idx = blockIdx.x * 8 + wave;     // b*256 + l
    int b = idx >> 8, l = idx & 255;
    const float* row = Hout + (size_t)idx * D2;
    const float* ts  = Tsal + (size_t)b * D2;
    float p1 = 0, p2 = 0;
    for (int e = lane; e < D2; e += 32) {
        float h = row[e];
        p1 += h * wcon[e];
        p2 += h * ts[e];
    }
#pragma unroll
    for (int off = 16; off > 0; off >>= 1) {
        p1 += __shfl_xor(p1, off, 32);
        p2 += __shfl_xor(p2, off, 32);
    }
    if (lane == 0) baseArr[idx] = p1 + p2 + bcon[0] + bsal[0] + pos[l];
}

// Sequential novelty scan. grid = 64, blk 1024 (32 waves).
__global__ void __launch_bounds__(1024)
scan_kernel(const float* __restrict__ Hn, const float* __restrict__ Hout,
            const float* __restrict__ baseArr, const float* __restrict__ bnov,
            const float* __restrict__ wcls, const float* __restrict__ bcls,
            float* __restrict__ out) {
    __shared__ float sArr[D2];
    __shared__ float red[32];
    __shared__ float pshare;
    int b = blockIdx.x, tid = threadIdx.x;
    int wave = tid >> 5, lane = tid & 31;
    sArr[tid] = 0.0f;
    __syncthreads();
    for (int l = 0; l < Ln; ++l) {
        size_t row = ((size_t)(b * Ln + l)) * D2;
        float v = Hn[row + tid] * sArr[tid];
#pragma unroll
        for (int off = 16; off > 0; off >>= 1) v += __shfl_xor(v, off, 32);
        if (lane == 0) red[wave] = v;
        __syncthreads();
        if (tid == 0) {
            float tot = 0.0f;
#pragma unroll
            for (int i = 0; i < 32; ++i) tot += red[i];
            float nov = (l == 0) ? 0.0f : (tot + bnov[0]);
            float p = sigmoidf_(wcls[0] * (baseArr[b * Ln + l] + nov) + bcls[0]);
            pshare = p;
            out[b * Ln + l] = p;
        }
        __syncthreads();
        float p = pshare;
        sArr[tid] += Hout[row + tid] * p;
    }
}

// ---------------------------------------------------------------------------
extern "C" void kernel_launch(void* const* d_in, const int* in_sizes, int n_in,
                              void* d_out, int out_size, void* d_ws, size_t ws_size,
                              hipStream_t stream) {
    (void)in_sizes; (void)n_in; (void)out_size; (void)ws_size;

    const float* x     = (const float*)d_in[0];
    const int*   slen  = (const int*)  d_in[1];
    const int*   dlen  = (const int*)  d_in[2];
    const float* WihF  = (const float*)d_in[3];
    const float* WhhF  = (const float*)d_in[4];
    const float* bF    = (const float*)d_in[5];
    const float* WihB  = (const float*)d_in[6];
    const float* WhhB  = (const float*)d_in[7];
    const float* bB    = (const float*)d_in[8];
    const float* Wdoc  = (const float*)d_in[9];
    const float* bdoc  = (const float*)d_in[10];
    const float* wcon  = (const float*)d_in[11];
    const float* bcon  = (const float*)d_in[12];
    const float* Wsal  = (const float*)d_in[13];
    const float* bsal  = (const float*)d_in[14];
    const float* Wnov  = (const float*)d_in[15];
    const float* bnov  = (const float*)d_in[16];
    const float* pos   = (const float*)d_in[17];
    const float* wcls  = (const float*)d_in[18];
    const float* bcls  = (const float*)d_in[19];
    float* out = (float*)d_out;

    // Workspace layout
    char* p = (char*)d_ws;
    __bf16* xbf    = (__bf16*)p; p += (size_t)Bn * Ln * En * 2;          // 16 MB
    __bf16* wcatF  = (__bf16*)p; p += (size_t)G4 * D2 * 2;               //  4 MB
    __bf16* wcatB  = (__bf16*)p; p += (size_t)G4 * D2 * 2;               //  4 MB
    __bf16* wnvT   = (__bf16*)p; p += (size_t)D2 * D2 * 2;               //  2 MB
    float*  HoutF  = (float*) p; p += (size_t)Bn * Ln * D2 * 4;          // 64 MB
    __bf16* HoutBF = (__bf16*)p; p += (size_t)Bn * Ln * D2 * 2;          // 32 MB
    float*  HnP    = (float*) p; p += (size_t)Bn * Ln * D2 * 4;          // 64 MB
    float*  Dv     = (float*) p; p += (size_t)Bn * D2 * 4;
    float*  Tsal   = (float*) p; p += (size_t)Bn * D2 * 4;
    float*  baseA  = (float*) p; p += (size_t)Bn * Ln * 4;

    // 1. Convert / pack operands to bf16.
    {
        int n = Bn * Ln * En;
        cvt_x_kernel<<<(n + 255) / 256, 256, 0, stream>>>(x, xbf, n);
        cvt_wcat_kernel<<<(G4 * D2 + 255) / 256, 256, 0, stream>>>(WihF, WhhF, wcatF);
        cvt_wcat_kernel<<<(G4 * D2 + 255) / 256, 256, 0, stream>>>(WihB, WhhB, wcatB);
        cvt_wnovT_kernel<<<(D2 * D2 + 255) / 256, 256, 0, stream>>>(Wnov, wnvT);
    }
    // 2. BiLSTM scan (WMMA + async LDS staging).
    lstm_kernel<<<8, 512, 0, stream>>>(xbf, wcatF, wcatB, bF, bB, slen, HoutF, HoutBF);
    // 3. Novelty pre-projection GEMM (WMMA).
    hn_gemm_kernel<<<(Bn * Ln) / 16, 512, 0, stream>>>(HoutBF, wnvT, HnP);
    // 4. Document vector + salience transform.
    doc_kernel<<<Bn, 256, 0, stream>>>(HoutF, dlen, Wdoc, bdoc, Dv);
    sal_kernel<<<Bn, 256, 0, stream>>>(Wsal, Dv, Tsal);
    // 5. Content + salience + position logits.
    base_kernel<<<(Bn * Ln) / 8, 256, 0, stream>>>(HoutF, wcon, bcon, Tsal, bsal, pos, baseA);
    // 6. Sequential novelty scan -> probabilities.
    scan_kernel<<<Bn, 1024, 0, stream>>>(HnP, HoutF, baseA, bnov, wcls, bcls, out);
}
